// rnn_one_20048907337988
// MI455X (gfx1250) — matrix operand order repaired and checked by reading, not compiled, not run on hardware
//
#include <hip/hip_runtime.h>

// Fused 2-layer GRU (B=16384, T=25, D=128) + output GEMM for gfx1250.
// f16 WMMA (v_wmma_f32_16x16x32_f16) with f32 accumulation.
// d_ws layout (filled by swizzle_weights each call, 425984 bytes needed):
//   [0      ) W1  swizzled f16, 96 KiB (96 blocks x 1 KiB)
//   [98304  ) U1  swizzled f16
//   [196608 ) W2  swizzled f16
//   [294912 ) U2  swizzled f16
//   [393216 ) Wout swizzled f16, 32 KiB
// LDS (320 KiB exactly): U1 | W2 | U2 | h1-state | h2-state

typedef __attribute__((ext_vector_type(16))) _Float16 v16h;
typedef __attribute__((ext_vector_type(8)))  float    v8f;

#define GRU_T 25
#define GRU_D 128

union V16U { uint4 u[2]; v16h h; };

__device__ __forceinline__ v8f wmma_f16(v16h a, v16h b, v8f c) {
  // 8 args: (neg_a, A, neg_b, B, c_mod, C, reuse_a, reuse_b)
  return __builtin_amdgcn_wmma_f32_16x16x32_f16(false, a, false, b, (short)0, c,
                                                false, false);
}

__device__ __forceinline__ v8f zero8() {
  v8f z;
#pragma unroll
  for (int i = 0; i < 8; ++i) z[i] = 0.0f;
  return z;
}

__device__ __forceinline__ float sigmoidf(float v) {
  return 1.0f / (1.0f + __expf(-v));
}

// Load one 16x16x32 f16 operand (A or B): 1 KiB block, 32 contiguous bytes/lane.
__device__ __forceinline__ v16h load_op(const char* base, int block, int lane) {
  const uint4* p = (const uint4*)(base + (block << 10) + (lane << 5));
  V16U u;
  u.u[0] = p[0];
  u.u[1] = p[1];
  return u.h;
}

// Scatter one f32 h-state element into the A-operand-layout buffer as f16.
// A 16x32 f16 layout: lane(0-15)=row, halves 0..7 -> K 0..7, halves 8..15 -> K 16..23;
// lanes 16-31 hold K 8..15 and 24..31 for the same rows.
__device__ __forceinline__ void store_hA(char* hbase, int m, int d, float v) {
  const int rt = m >> 4, ks = d >> 5, r = m & 15, kk = d & 31;
  const int ln = r + (((kk >> 3) & 1) << 4);
  const int hf = (kk & 7) + ((kk >> 4) << 3);
  *(_Float16*)(hbase + ((rt * 4 + ks) << 10) + (ln << 5) + (hf << 1)) = (_Float16)v;
}

// Build an A operand for x_t straight from global f32 (two 8-wide K runs per lane).
__device__ __forceinline__ v16h load_xA(const float* xp, int kb) {
  float4 f0 = *(const float4*)(xp + kb);
  float4 f1 = *(const float4*)(xp + kb + 4);
  float4 f2 = *(const float4*)(xp + kb + 16);
  float4 f3 = *(const float4*)(xp + kb + 20);
  v16h a;
  a[0] = (_Float16)f0.x; a[1] = (_Float16)f0.y; a[2] = (_Float16)f0.z; a[3] = (_Float16)f0.w;
  a[4] = (_Float16)f1.x; a[5] = (_Float16)f1.y; a[6] = (_Float16)f1.z; a[7] = (_Float16)f1.w;
  a[8] = (_Float16)f2.x; a[9] = (_Float16)f2.y; a[10] = (_Float16)f2.z; a[11] = (_Float16)f2.w;
  a[12] = (_Float16)f3.x; a[13] = (_Float16)f3.y; a[14] = (_Float16)f3.z; a[15] = (_Float16)f3.w;
  return a;
}

// Convert f32 weight matrix [128, N] (rows = K) into swizzled f16 B-operand blocks.
// One block per (nt, ks): lane<16 -> col nt*16+lane, K ks*32+0..7,16..23; lane>=16 -> K 8..15,24..31.
__global__ void swizzle_weights(const float* __restrict__ src, char* __restrict__ dst,
                                int N) {
  const int blk = blockIdx.x;          // nt*4 + ks
  const int nt = blk >> 2, ks = blk & 3;
  const int lane = threadIdx.x;
  const int col = nt * 16 + (lane & 15);
  const int kb = ks * 32 + (lane >> 4) * 8;
  _Float16* d = (_Float16*)(dst + (blk << 10) + (lane << 5));
#pragma unroll
  for (int h = 0; h < 16; ++h)
    d[h] = (_Float16)src[(size_t)(kb + h + (h >> 3) * 8) * N + col];
}

__global__ __launch_bounds__(256, 1) void gru2_fused(const float* __restrict__ x,
                                                     const char* __restrict__ wsw,
                                                     float* __restrict__ out) {
  __shared__ __align__(16) char smem[327680];
  char* U1s = smem;                 // 98304
  char* W2s = smem + 98304;         // 98304
  char* U2s = smem + 196608;        // 98304
  char* H1  = smem + 294912;        // 16384 (A-operand layout, 4 rt x 4 ks blocks)
  char* H2  = smem + 311296;        // 16384

  const int tid  = threadIdx.x;
  const int wave = tid >> 5;        // 0..7  -> owns d-tile `wave`
  const int lane = tid & 31;
  const int hi   = lane >> 4;
  const int ln16 = lane & 15;
  const int b0   = blockIdx.x * 64; // 64 batch rows per workgroup

  // Stage U1/W2/U2 (contiguous 294912 B in ws at +98304) into LDS; zero h states.
  {
    const uint4* s = (const uint4*)(wsw + 98304);
    uint4* d = (uint4*)smem;
    for (int i = tid; i < 294912 / 16; i += 256) d[i] = s[i];
    uint4 z; z.x = z.y = z.z = z.w = 0u;
    uint4* hz = (uint4*)H1;
    for (int i = tid; i < 32768 / 16; i += 256) hz[i] = z;
  }
  __syncthreads();

  // 32-bit element offsets for this thread's x rows (keeps addressing SGPR+u32).
  unsigned xoff[4];
#pragma unroll
  for (int rt = 0; rt < 4; ++rt)
    xoff[rt] = (unsigned)(b0 + rt * 16 + ln16) * (unsigned)(GRU_T * GRU_D);

  v8f hp1[4], hp2[4];               // carried states (f32) for rows rt, cols d-tile `wave`
#pragma unroll
  for (int rt = 0; rt < 4; ++rt) { hp1[rt] = zero8(); hp2[rt] = zero8(); }

  // Accumulators: z and r gates take xW + hU summed (shared C chain);
  // candidate gate needs xh and uh separately (reset_after).
  v8f azr[2][4], axh[4], auh[4];

#pragma unroll 1
  for (int t = 0; t < GRU_T; ++t) {
    const unsigned tOff = (unsigned)t * GRU_D;

    // ---------------- layer 1: azr = x@W1zr + h1@U1zr ; axh = x@W1h ; auh = h1@U1h
#pragma unroll
    for (int rt = 0; rt < 4; ++rt) {
      azr[0][rt] = zero8(); azr[1][rt] = zero8();
      axh[rt] = zero8();    auh[rt] = zero8();
    }

#pragma unroll
    for (int ks = 0; ks < 4; ++ks) {
      const int kb = ks * 32 + hi * 8;
      v16h Ax[4], Ah[4];
#pragma unroll
      for (int rt = 0; rt < 4; ++rt) {
        Ax[rt] = load_xA(x + xoff[rt] + tOff, kb);
        Ah[rt] = load_op(H1, rt * 4 + ks, lane);
      }
#pragma unroll
      for (int g = 0; g < 2; ++g) {                    // z, r gates: summed C
        v16h Bw = load_op(wsw, (wave + 8 * g) * 4 + ks, lane);  // W1 via L2
        v16h Bu = load_op(U1s, (wave + 8 * g) * 4 + ks, lane);  // U1 in LDS
#pragma unroll
        for (int rt = 0; rt < 4; ++rt) {
          azr[g][rt] = wmma_f16(Ax[rt], Bw, azr[g][rt]);
          azr[g][rt] = wmma_f16(Ah[rt], Bu, azr[g][rt]);
        }
      }
      {                                                // candidate: separate C
        v16h Bw = load_op(wsw, (wave + 16) * 4 + ks, lane);
        v16h Bu = load_op(U1s, (wave + 16) * 4 + ks, lane);
#pragma unroll
        for (int rt = 0; rt < 4; ++rt) {
          axh[rt] = wmma_f16(Ax[rt], Bw, axh[rt]);
          auh[rt] = wmma_f16(Ah[rt], Bu, auh[rt]);
        }
      }
    }
    __syncthreads();                                  // all H1 reads done

#pragma unroll
    for (int rt = 0; rt < 4; ++rt) {                  // gates, update h1
      v8f hn;
#pragma unroll
      for (int i = 0; i < 8; ++i) {
        float z  = sigmoidf(azr[0][rt][i]);
        float r  = sigmoidf(azr[1][rt][i]);
        float hh = axh[rt][i] + r * auh[rt][i];       // reset_after, linear act
        hn[i] = z * hp1[rt][i] + (1.0f - z) * hh;
      }
      hp1[rt] = hn;
      const int d = wave * 16 + ln16;
#pragma unroll
      for (int v = 0; v < 8; ++v) store_hA(H1, rt * 16 + v + 8 * hi, d, hn[v]);
    }
    __syncthreads();                                  // H1 (= h1_t) visible

    // ---------------- layer 2: azr = h1@W2zr + h2@U2zr ; axh = h1@W2h ; auh = h2@U2h
#pragma unroll
    for (int rt = 0; rt < 4; ++rt) {
      azr[0][rt] = zero8(); azr[1][rt] = zero8();
      axh[rt] = zero8();    auh[rt] = zero8();
    }

#pragma unroll
    for (int ks = 0; ks < 4; ++ks) {
      v16h A1[4], A2[4];
#pragma unroll
      for (int rt = 0; rt < 4; ++rt) {
        A1[rt] = load_op(H1, rt * 4 + ks, lane);
        A2[rt] = load_op(H2, rt * 4 + ks, lane);
      }
#pragma unroll
      for (int g = 0; g < 2; ++g) {
        v16h Bw = load_op(W2s, (wave + 8 * g) * 4 + ks, lane);
        v16h Bu = load_op(U2s, (wave + 8 * g) * 4 + ks, lane);
#pragma unroll
        for (int rt = 0; rt < 4; ++rt) {
          azr[g][rt] = wmma_f16(A1[rt], Bw, azr[g][rt]);
          azr[g][rt] = wmma_f16(A2[rt], Bu, azr[g][rt]);
        }
      }
      {
        v16h Bw = load_op(W2s, (wave + 16) * 4 + ks, lane);
        v16h Bu = load_op(U2s, (wave + 16) * 4 + ks, lane);
#pragma unroll
        for (int rt = 0; rt < 4; ++rt) {
          axh[rt] = wmma_f16(A1[rt], Bw, axh[rt]);
          auh[rt] = wmma_f16(A2[rt], Bu, auh[rt]);
        }
      }
    }
    __syncthreads();                                  // all H2 reads done

#pragma unroll
    for (int rt = 0; rt < 4; ++rt) {                  // gates, update h2
      v8f hn;
#pragma unroll
      for (int i = 0; i < 8; ++i) {
        float z  = sigmoidf(azr[0][rt][i]);
        float r  = sigmoidf(azr[1][rt][i]);
        float hh = axh[rt][i] + r * auh[rt][i];
        hn[i] = z * hp2[rt][i] + (1.0f - z) * hh;
      }
      hp2[rt] = hn;
      const int d = wave * 16 + ln16;
#pragma unroll
      for (int v = 0; v < 8; ++v) store_hA(H2, rt * 16 + v + 8 * hi, d, hn[v]);
    }
    __syncthreads();                                  // H2 (= h2_t) visible
  }

  // ---------------- out = h2_T @ Wout ; wave w handles n-tile w ----------------
  v8f ao[4];
#pragma unroll
  for (int rt = 0; rt < 4; ++rt) ao[rt] = zero8();
#pragma unroll
  for (int ks = 0; ks < 4; ++ks) {
    v16h Bo = load_op(wsw + 393216, wave * 4 + ks, lane);     // Wout via L2
#pragma unroll
    for (int rt = 0; rt < 4; ++rt) {
      v16h A = load_op(H2, rt * 4 + ks, lane);
      ao[rt] = wmma_f16(A, Bo, ao[rt]);
    }
  }
#pragma unroll
  for (int rt = 0; rt < 4; ++rt)
#pragma unroll
    for (int v = 0; v < 8; ++v)
      out[(size_t)(b0 + rt * 16 + v + 8 * hi) * GRU_D + wave * 16 + ln16] = ao[rt][v];
}

extern "C" void kernel_launch(void* const* d_in, const int* in_sizes, int n_in,
                              void* d_out, int out_size, void* d_ws, size_t ws_size,
                              hipStream_t stream) {
  const float* x  = (const float*)d_in[0];
  const float* W1 = (const float*)d_in[1];
  const float* U1 = (const float*)d_in[2];
  const float* W2 = (const float*)d_in[3];
  const float* U2 = (const float*)d_in[4];
  const float* Wo = (const float*)d_in[5];
  float* out = (float*)d_out;
  char* ws = (char*)d_ws;
  const int B = in_sizes[0] / (GRU_T * GRU_D);   // 16384

  swizzle_weights<<<96, 32, 0, stream>>>(W1, ws + 0,      3 * GRU_D);
  swizzle_weights<<<96, 32, 0, stream>>>(U1, ws + 98304,  3 * GRU_D);
  swizzle_weights<<<96, 32, 0, stream>>>(W2, ws + 196608, 3 * GRU_D);
  swizzle_weights<<<96, 32, 0, stream>>>(U2, ws + 294912, 3 * GRU_D);
  swizzle_weights<<<32, 32, 0, stream>>>(Wo, ws + 393216, GRU_D);

  gru2_fused<<<B / 64, 256, 0, stream>>>(x, ws, out);

  (void)n_in; (void)out_size; (void)ws_size;
}
